// HiGCN_4406636446289
// MI455X (gfx1250) — matrix operand order repaired and checked by reading, not compiled, not run on hardware
//
#include <hip/hip_runtime.h>
#include <hip/hip_bf16.h>

#define SLOPE 0.01f
#define EPSV  1e-5f

typedef __bf16 bf16_t;
typedef __attribute__((ext_vector_type(8)))  bf16_t v8bf;
typedef __attribute__((ext_vector_type(16))) bf16_t v16bf;
typedef __attribute__((ext_vector_type(8)))  float  v8f;

__device__ __forceinline__ unsigned short f2bf(float f) {
  unsigned u = __float_as_uint(f);
  u += 0x7FFFu + ((u >> 16) & 1u);
  return (unsigned short)(u >> 16);
}

__device__ __forceinline__ float leaky(float v) {
  return v >= 0.f ? v : SLOPE * v;
}

// ---------------------------------------------------------------------------
// Per-column (feature/channel) batch-norm statistics of X[N,F] (training-mode,
// biased variance). Emits scale/shift so that BN(x) = x*sc[f] + sh[f].
// One block per feature column.
// ---------------------------------------------------------------------------
__global__ void colstats_k(const float* __restrict__ X, int N, int F,
                           const float* __restrict__ g, const float* __restrict__ b,
                           float* __restrict__ sc, float* __restrict__ sh) {
  __shared__ float s1[256], s2[256];
  int f = blockIdx.x;
  int tid = threadIdx.x;
  float a = 0.f, q = 0.f;
  for (int i = tid; i < N; i += 256) {
    float v = X[(size_t)i * F + f];
    a += v; q += v * v;
  }
  s1[tid] = a; s2[tid] = q;
  __syncthreads();
  for (int o = 128; o > 0; o >>= 1) {
    if (tid < o) { s1[tid] += s1[tid + o]; s2[tid] += s2[tid + o]; }
    __syncthreads();
  }
  if (tid == 0) {
    float mu  = s1[0] / (float)N;
    float var = s2[0] / (float)N - mu * mu;
    float r   = rsqrtf(var + EPSV);
    float scale = g[f] * r;
    sc[f] = scale;
    sh[f] = b[f] - mu * scale;
  }
}

// Fused BN-normalize + f32->bf16 convert into padded WMMA A operand [N,Kpad].
__global__ void bnorm_bf16_k(const float* __restrict__ X,
                             const float* __restrict__ sc, const float* __restrict__ sh,
                             unsigned short* __restrict__ A, int N, int F, int Kpad) {
  size_t i = (size_t)blockIdx.x * blockDim.x + threadIdx.x;
  size_t total = (size_t)N * Kpad;
  if (i >= total) return;
  int k = (int)(i % Kpad);
  size_t n = i / Kpad;
  float v = (k < F) ? (X[n * F + k] * sc[k] + sh[k]) : 0.f;
  A[i] = f2bf(v);
}

// Row-major f32 [N,K] -> bf16 [N,Kpad] (zero padded).
__global__ void cvt_row_bf16_k(const float* __restrict__ X,
                               unsigned short* __restrict__ A,
                               int N, int K, int Kpad) {
  size_t i = (size_t)blockIdx.x * blockDim.x + threadIdx.x;
  size_t total = (size_t)N * Kpad;
  if (i >= total) return;
  int k = (int)(i % Kpad);
  size_t n = i / Kpad;
  A[i] = f2bf((k < K) ? X[n * K + k] : 0.f);
}

// Transpose-convert f32 W[K,N] -> bf16 Bt[N,Kpad] (zero padded).
__global__ void cvt_tr_bf16_k(const float* __restrict__ W,
                              unsigned short* __restrict__ Bt,
                              int K, int N, int Kpad) {
  size_t i = (size_t)blockIdx.x * blockDim.x + threadIdx.x;
  size_t total = (size_t)N * Kpad;
  if (i >= total) return;
  int k = (int)(i % Kpad);
  size_t n = i / Kpad;
  Bt[i] = f2bf((k < K) ? W[(size_t)k * N + n] : 0.f);
}

// ---------------------------------------------------------------------------
// WMMA GEMM v2: C[M,N] = A[M,Kpad](bf16) * Bt[N,Kpad](bf16)^T + bias[N],
// optional LeakyReLU.
//  - One block (4 waves) per 16-row M block. The 16xKpad A block (contiguous
//    in memory) is staged once into LDS (global_load_b128 + ds_store_b128) and
//    shared by all waves; fragment reads come from LDS (ds_load_b128).
//  - Each wave owns NT adjacent 16-wide N tiles and keeps NT f32 accumulators,
//    reusing the A fragment across NT wmmas per K step.
// Fragment layouts used here:
//  A (16x32 bf16): lane L (row L&15) holds 16 contiguous K starting 16*(L>=16).
//  B (32x16 bf16): lane L (col L&15) holds 16 contiguous K starting 16*(L>=16).
//  C/D (16x16 f32): element (m,n): m = v + 8*(L>=16), n = L&15.
// ---------------------------------------------------------------------------
template <int NT>
__global__ __launch_bounds__(128) void wmma_gemm2_k(
    const unsigned short* __restrict__ A, const unsigned short* __restrict__ Bt,
    const float* __restrict__ bias, float* __restrict__ C,
    int M, int N, int Kpad, int do_leaky) {
  __shared__ __align__(16) unsigned short As[16 * 224];   // up to Kpad=224
  int tid  = threadIdx.x;
  int wave = tid >> 5;
  int lane = tid & 31;
  int tm   = blockIdx.x;

  // ---- stage the 16-row A block (flat 16*Kpad bf16 chunk) into LDS
  {
    const uint4* src = (const uint4*)(A + (size_t)tm * 16 * Kpad);
    uint4* dst = (uint4*)As;
    int chunks = (16 * Kpad) >> 3;          // 8 bf16 per uint4
    for (int i = tid; i < chunks; i += 128) dst[i] = src[i];
  }
  __syncthreads();

  int ntile  = N >> 4;
  int nwaves = ntile / NT;                  // waves that have work
  if (wave < nwaves) {                      // wave-uniform: EXEC stays all-1s
    int l16 = lane & 15;
    int hi  = lane >> 4;
    const bf16_t* Al = (const bf16_t*)As + (size_t)l16 * Kpad + hi * 16;

    const bf16_t* Br[NT];
    v8f c[NT];
#pragma unroll
    for (int t = 0; t < NT; ++t) {
      int tn = wave * NT + t;
      Br[t] = (const bf16_t*)Bt + (size_t)(tn * 16 + l16) * Kpad + hi * 16;
      __builtin_prefetch(Br[t], 0, 1);      // global_prefetch_b8 (L2-resident weights)
      float bval = bias[tn * 16 + l16];
#pragma unroll
      for (int v = 0; v < 8; ++v) c[t][v] = bval;
    }

    for (int k0 = 0; k0 < Kpad; k0 += 32) {
      union { v16bf v; v8bf h[2]; } a;
      a.h[0] = *(const v8bf*)(Al + k0);         // ds_load_b128
      a.h[1] = *(const v8bf*)(Al + k0 + 8);
#pragma unroll
      for (int t = 0; t < NT; ++t) {
        v16bf bm = *(const v16bf*)(Br[t] + k0); // global_load_b128 x2
        c[t] = __builtin_amdgcn_wmma_f32_16x16x32_bf16(
            /*neg_a=*/false, a.v, /*neg_b=*/false, bm,
            /*c_mod=*/(short)0, c[t], /*reuse_a=*/false, /*reuse_b=*/false);
      }
    }

#pragma unroll
    for (int t = 0; t < NT; ++t) {
      int tn = wave * NT + t;
      float* Cp = C + ((size_t)(tm * 16 + hi * 8)) * N + tn * 16 + l16;
#pragma unroll
      for (int v = 0; v < 8; ++v) {
        float val = c[t][v];
        if (do_leaky) val = leaky(val);
        Cp[(size_t)v * N] = val;
      }
    }
  }
}

// ---------------------------------------------------------------------------
// Depthwise 5x5 SAME conv over 128x128 spatial, 128 channels, HWC layout
// (pixel-major [h*128+w][128]), + bias + LeakyReLU. Thread per (pixel,ch).
// ---------------------------------------------------------------------------
__global__ void dwconv5_k(const float* __restrict__ in, const float* __restrict__ w,
                          const float* __restrict__ wb, float* __restrict__ out) {
  int idx = blockIdx.x * blockDim.x + threadIdx.x;    // 16384*128
  int ch = idx & 127;
  int p  = idx >> 7;
  int y  = p >> 7, x = p & 127;
  float acc = wb[ch];
#pragma unroll
  for (int dy = -2; dy <= 2; ++dy) {
    int yy = y + dy;
    if (yy < 0 || yy > 127) continue;
#pragma unroll
    for (int dx = -2; dx <= 2; ++dx) {
      int xx = x + dx;
      if (xx < 0 || xx > 127) continue;
      acc += in[(size_t)((yy << 7) + xx) * 128 + ch] * w[ch * 25 + (dy + 2) * 5 + (dx + 2)];
    }
  }
  out[idx] = leaky(acc);
}

// Pool groups of 4 consecutive rows (Shat @ H with balanced assignment).
__global__ void pool4_k(const float* __restrict__ in, float* __restrict__ out,
                        int Nout, int F) {
  int idx = blockIdx.x * blockDim.x + threadIdx.x;
  if (idx >= Nout * F) return;
  int f = idx % F;
  int i = idx / F;
  const float* p = in + (size_t)(i * 4) * F + f;
  out[idx] = 0.25f * (p[0] + p[F] + p[2 * F] + p[3 * F]);
}

// out[i] = concat(up[i/4][0:F1], cat[i][0:F2])  (unpool-by-4 then concat).
__global__ void unpool_cat_k(const float* __restrict__ up, const float* __restrict__ cat,
                             float* __restrict__ out, int Nout, int F1, int F2) {
  int Ft = F1 + F2;
  int idx = blockIdx.x * blockDim.x + threadIdx.x;
  if (idx >= Nout * Ft) return;
  int f = idx % Ft;
  int i = idx / Ft;
  out[idx] = (f < F1) ? up[(size_t)(i >> 2) * F1 + f]
                      : cat[(size_t)i * F2 + (f - F1)];
}

// ---------------------------------------------------------------------------
// GCN attention: per node n, A_k = softmax_k(sigmoid(<th[nbr[n,k]], th[n]>)),
// out = leaky(sum_k A_k * outF[nbr[n,k]]). One wave32 per node; lanes own
// neighbors for the dot/softmax, then own output features for the aggregate.
// th feature dim fixed at 128; Fout in {32, 64}.
// ---------------------------------------------------------------------------
__global__ void gcn_attn_k(const float* __restrict__ th, const float* __restrict__ outF,
                           const int* __restrict__ nbr, float* __restrict__ dst,
                           int N, int Fout) {
  int node = blockIdx.x * (blockDim.x >> 5) + (threadIdx.x >> 5);
  int lane = threadIdx.x & 31;
  if (node >= N) return;
  int k = lane & 15;                       // lanes 16-31 mirror lanes 0-15
  int m = nbr[node * 16 + k];
  const float* tn = th + (size_t)node * 128;
  const float* tm = th + (size_t)m * 128;
  float s = 0.f;
  for (int f = 0; f < 128; ++f) s += tn[f] * tm[f];
  float a = 1.f / (1.f + __expf(-s));      // sigmoid
  // softmax over the 16 neighbors (xor-reduce within 16-lane halves)
  float mx = a;
  for (int o = 8; o >= 1; o >>= 1) mx = fmaxf(mx, __shfl_xor(mx, o, 32));
  float e = __expf(a - mx);
  float ssum = e;
  for (int o = 8; o >= 1; o >>= 1) ssum += __shfl_xor(ssum, o, 32);
  float w = e / ssum;
  // weighted aggregate: lane owns features {lane, lane+32}
  float acc0 = 0.f, acc1 = 0.f;
  for (int kk = 0; kk < 16; ++kk) {
    float wk = __shfl(w, kk, 32);
    int   mk = __shfl(m, kk, 32);
    const float* om = outF + (size_t)mk * Fout;
    if (lane < Fout)      acc0 += wk * om[lane];
    if (lane + 32 < Fout) acc1 += wk * om[lane + 32];
  }
  if (lane < Fout)      dst[(size_t)node * Fout + lane]      = leaky(acc0);
  if (lane + 32 < Fout) dst[(size_t)node * Fout + lane + 32] = leaky(acc1);
}

// Row-wise softmax over 16 classes; thread per row.
__global__ void softmax16_k(const float* __restrict__ in, float* __restrict__ out, int N) {
  int i = blockIdx.x * blockDim.x + threadIdx.x;
  if (i >= N) return;
  const float* r = in + (size_t)i * 16;
  float mx = r[0];
#pragma unroll
  for (int j = 1; j < 16; ++j) mx = fmaxf(mx, r[j]);
  float e[16], s = 0.f;
#pragma unroll
  for (int j = 0; j < 16; ++j) { e[j] = __expf(r[j] - mx); s += e[j]; }
  float inv = 1.f / s;
#pragma unroll
  for (int j = 0; j < 16; ++j) out[(size_t)i * 16 + j] = e[j] * inv;
}

// ---------------------------------------------------------------------------
// Host-side orchestration
// ---------------------------------------------------------------------------
static inline int cdiv(long long a, long long b) { return (int)((a + b - 1) / b); }

extern "C" void kernel_launch(void* const* d_in, const int* in_sizes, int n_in,
                              void* d_out, int out_size, void* d_ws, size_t ws_size,
                              hipStream_t stream) {
  (void)in_sizes; (void)n_in; (void)out_size; (void)ws_size;
  // setup_inputs() dict order
  const float* x     = (const float*)d_in[0];   // [16384, 200]
  const int*   nbr_a = (const int*)  d_in[5];   // [4096, 16]
  const int*   nbr_b = (const int*)  d_in[6];   // [1024, 16]
  const float* hg  = (const float*)d_in[7];
  const float* hb  = (const float*)d_in[8];
  const float* hpw = (const float*)d_in[9];     // [128, 200]
  const float* hpb = (const float*)d_in[10];
  const float* hdw = (const float*)d_in[11];    // [128, 25]
  const float* hdb = (const float*)d_in[12];
  const float* e0_g = (const float*)d_in[13]; const float* e0_b = (const float*)d_in[14];
  const float* e0_wt = (const float*)d_in[15]; const float* e0_bt = (const float*)d_in[16];
  const float* e0_wo = (const float*)d_in[17]; const float* e0_bo = (const float*)d_in[18];
  const float* e1_g = (const float*)d_in[19]; const float* e1_b = (const float*)d_in[20];
  const float* e1_wt = (const float*)d_in[21]; const float* e1_bt = (const float*)d_in[22];
  const float* e1_wo = (const float*)d_in[23]; const float* e1_bo = (const float*)d_in[24];
  const float* d0_g = (const float*)d_in[25]; const float* d0_b = (const float*)d_in[26];
  const float* d0_wt = (const float*)d_in[27]; const float* d0_bt = (const float*)d_in[28];
  const float* d0_wo = (const float*)d_in[29]; const float* d0_bo = (const float*)d_in[30];
  const float* tg  = (const float*)d_in[31];
  const float* tb  = (const float*)d_in[32];
  const float* tpw = (const float*)d_in[33];    // [128, 192]
  const float* tpb = (const float*)d_in[34];
  const float* tdw = (const float*)d_in[35];
  const float* tdb = (const float*)d_in[36];
  const float* wsm = (const float*)d_in[37];    // [128, 16]
  const float* wbv = (const float*)d_in[38];

  const int N0 = 16384, N1 = 4096, N2 = 1024;

  // bump allocator over workspace
  char* base = (char*)d_ws;
  size_t off = 0;
  auto alloc = [&](size_t bytes) -> char* {
    char* p = base + off;
    off += (bytes + 255) & ~(size_t)255;
    return p;
  };
  float* sc  = (float*)alloc(256 * 4);
  float* sh  = (float*)alloc(256 * 4);
  unsigned short* bfA = (unsigned short*)alloc((size_t)N0 * 224 * 2); // A operand (reused)
  unsigned short* bfB = (unsigned short*)alloc(32768 * 2);            // Bt operand (reused)
  float* P1   = (float*)alloc((size_t)N0 * 128 * 4); // head/tail pointwise output (reused)
  float* H0   = (float*)alloc((size_t)N0 * 128 * 4); // enc0; reused for final features
  float* Hp   = (float*)alloc((size_t)N1 * 128 * 4); // pooled level-1
  float* th   = (float*)alloc((size_t)N1 * 128 * 4); // theta (reused per GCN)
  float* oF   = (float*)alloc((size_t)N1 * 64 * 4);  // out features (reused per GCN)
  float* He0  = (float*)alloc((size_t)N1 * 64 * 4);  // enc1
  float* Hp2  = (float*)alloc((size_t)N2 * 64 * 4);
  float* He1  = (float*)alloc((size_t)N2 * 32 * 4);
  float* Hcat = (float*)alloc((size_t)N1 * 96 * 4);
  float* Hd0  = (float*)alloc((size_t)N1 * 64 * 4);
  float* Tcat = (float*)alloc((size_t)N0 * 192 * 4);
  float* logits = (float*)alloc((size_t)N0 * 16 * 4);

  auto gemm = [&](const unsigned short* A, const unsigned short* Bt, const float* bias,
                  float* C, int M, int N, int Kpad, int lk) {
    int mblocks = M / 16;
    if ((N / 16) >= 8) {
      wmma_gemm2_k<2><<<mblocks, 128, 0, stream>>>(A, Bt, bias, C, M, N, Kpad, lk);
    } else {
      wmma_gemm2_k<1><<<mblocks, 128, 0, stream>>>(A, Bt, bias, C, M, N, Kpad, lk);
    }
  };

  // ---- head SSConv: BN2d(200) -> 1x1 conv (200->128) -> leaky -> dw5x5 -> leaky
  colstats_k<<<200, 256, 0, stream>>>(x, N0, 200, hg, hb, sc, sh);
  bnorm_bf16_k<<<cdiv((long long)N0 * 224, 256), 256, 0, stream>>>(x, sc, sh, bfA, N0, 200, 224);
  cvt_row_bf16_k<<<cdiv(128 * 224, 256), 256, 0, stream>>>(hpw, bfB, 128, 200, 224);
  gemm(bfA, bfB, hpb, P1, N0, 128, 224, 1);
  dwconv5_k<<<cdiv((long long)N0 * 128, 256), 256, 0, stream>>>(P1, hdw, hdb, H0); // enc0
  pool4_k<<<cdiv((long long)N1 * 128, 256), 256, 0, stream>>>(H0, Hp, N1, 128);

  // ---- GCN e0 (128 -> 64) on N1 nodes, nbr_a
  colstats_k<<<128, 256, 0, stream>>>(Hp, N1, 128, e0_g, e0_b, sc, sh);
  bnorm_bf16_k<<<cdiv((long long)N1 * 128, 256), 256, 0, stream>>>(Hp, sc, sh, bfA, N1, 128, 128);
  cvt_tr_bf16_k<<<cdiv(128 * 128, 256), 256, 0, stream>>>(e0_wt, bfB, 128, 128, 128);
  gemm(bfA, bfB, e0_bt, th, N1, 128, 128, 0);
  cvt_tr_bf16_k<<<cdiv(64 * 128, 256), 256, 0, stream>>>(e0_wo, bfB, 128, 64, 128);
  gemm(bfA, bfB, e0_bo, oF, N1, 64, 128, 0);
  gcn_attn_k<<<cdiv(N1, 8), 256, 0, stream>>>(th, oF, nbr_a, He0, N1, 64); // enc1
  pool4_k<<<cdiv((long long)N2 * 64, 256), 256, 0, stream>>>(He0, Hp2, N2, 64);

  // ---- GCN e1 (64 -> 32) on N2 nodes, nbr_b
  colstats_k<<<64, 256, 0, stream>>>(Hp2, N2, 64, e1_g, e1_b, sc, sh);
  bnorm_bf16_k<<<cdiv((long long)N2 * 64, 256), 256, 0, stream>>>(Hp2, sc, sh, bfA, N2, 64, 64);
  cvt_tr_bf16_k<<<cdiv(128 * 64, 256), 256, 0, stream>>>(e1_wt, bfB, 64, 128, 64);
  gemm(bfA, bfB, e1_bt, th, N2, 128, 64, 0);
  cvt_tr_bf16_k<<<cdiv(32 * 64, 256), 256, 0, stream>>>(e1_wo, bfB, 64, 32, 64);
  gemm(bfA, bfB, e1_bo, oF, N2, 32, 64, 0);
  gcn_attn_k<<<cdiv(N2, 8), 256, 0, stream>>>(th, oF, nbr_b, He1, N2, 32);

  // ---- decoder GCN d0 (96 -> 64): unpool He1 + concat enc1
  unpool_cat_k<<<cdiv((long long)N1 * 96, 256), 256, 0, stream>>>(He1, He0, Hcat, N1, 32, 64);
  colstats_k<<<96, 256, 0, stream>>>(Hcat, N1, 96, d0_g, d0_b, sc, sh);
  bnorm_bf16_k<<<cdiv((long long)N1 * 128, 256), 256, 0, stream>>>(Hcat, sc, sh, bfA, N1, 96, 128);
  cvt_tr_bf16_k<<<cdiv(128 * 128, 256), 256, 0, stream>>>(d0_wt, bfB, 96, 128, 128);
  gemm(bfA, bfB, d0_bt, th, N1, 128, 128, 0);
  cvt_tr_bf16_k<<<cdiv(64 * 128, 256), 256, 0, stream>>>(d0_wo, bfB, 96, 64, 128);
  gemm(bfA, bfB, d0_bo, oF, N1, 64, 128, 0);
  gcn_attn_k<<<cdiv(N1, 8), 256, 0, stream>>>(th, oF, nbr_a, Hd0, N1, 64);

  // ---- tail SSConv on concat(unpool(Hd0), enc0) = [N0, 192]
  unpool_cat_k<<<cdiv((long long)N0 * 192, 256), 256, 0, stream>>>(Hd0, H0, Tcat, N0, 64, 128);
  colstats_k<<<192, 256, 0, stream>>>(Tcat, N0, 192, tg, tb, sc, sh);
  bnorm_bf16_k<<<cdiv((long long)N0 * 192, 256), 256, 0, stream>>>(Tcat, sc, sh, bfA, N0, 192, 192);
  cvt_row_bf16_k<<<cdiv(128 * 192, 256), 256, 0, stream>>>(tpw, bfB, 128, 192, 192);
  gemm(bfA, bfB, tpb, P1, N0, 128, 192, 1);
  dwconv5_k<<<cdiv((long long)N0 * 128, 256), 256, 0, stream>>>(P1, tdw, tdb, H0); // final feats

  // ---- classifier: softmax(F @ ws + wb)
  cvt_row_bf16_k<<<cdiv((long long)N0 * 128, 256), 256, 0, stream>>>(H0, bfA, N0, 128, 128);
  cvt_tr_bf16_k<<<cdiv(16 * 128, 256), 256, 0, stream>>>(wsm, bfB, 128, 16, 128);
  gemm(bfA, bfB, wbv, logits, N0, 16, 128, 0);
  softmax16_k<<<cdiv(N0, 256), 256, 0, stream>>>(logits, (float*)d_out, N0);
}
